// D4RTEncoder_25623774888524
// MI455X (gfx1250) — matrix operand order repaired and checked by reading, not compiled, not run on hardware
//
#include <hip/hip_runtime.h>
#include <hip/hip_bf16.h>
#include <math.h>

// ---------------- types for WMMA ----------------
typedef __attribute__((ext_vector_type(16))) __bf16         v16bf;
typedef __attribute__((ext_vector_type(16))) unsigned short v16u;
typedef __attribute__((ext_vector_type(8)))  float          v8f;

// ---------------- model constants ----------------
#define CDIM   768
#define SEQ    257
#define BATCH  4
#define MROWS  (BATCH * SEQ)   // 1028
#define MPAD   1040            // 65 * 16
#define C3     (3 * CDIM)      // 2304
#define CH     3351
#define CHP    3360            // 105 * 32 = 35 * 96
#define NHEADS 12
#define HD     64
#define KCHUNK 96              // LDS staging chunk (3 WMMA k-steps)

// ---------------- helpers ----------------
__device__ __forceinline__ unsigned short f2bf_u(float f) {
  unsigned u = __builtin_bit_cast(unsigned, f);
  unsigned r = u + 0x7FFFu + ((u >> 16) & 1u);   // round-to-nearest-even
  return (unsigned short)(r >> 16);
}

__device__ __forceinline__ float wave_sum32(float v) {
#pragma unroll
  for (int off = 16; off > 0; off >>= 1) v += __shfl_xor(v, off, 32);
  return v;
}

__device__ __forceinline__ float block_sum256(float v, float* red) {
  int t = threadIdx.x;
  red[t] = v;
  __syncthreads();
#pragma unroll
  for (int off = 128; off > 0; off >>= 1) {
    if (t < off) red[t] += red[t + off];
    __syncthreads();
  }
  float r = red[0];
  __syncthreads();
  return r;
}

__device__ __forceinline__ float gelu_exact(float x) {
  return 0.5f * x * (1.0f + erff(x * 0.70710678118654752f));
}

// =========================================================================
// WMMA GEMM:  out[M x N] = A_bf16[Mpad x Kpad] * W_f32[N x K]^T (+ bias)
//
// Block = 8 waves covering a 16(M) x 128(N) strip.
// The shared 16 x KCHUNK A tile is staged into LDS once per block with
// global_load_async_to_lds_b128 (ASYNCcnt), double-buffered; each wave
// builds its A fragment from LDS and streams its own fp32 weight rows,
// converting to bf16 in-register for v_wmma_f32_16x16x32_bf16.
// A is zero-padded (rows >= M, cols >= K), so B reads are clamped in-range
// instead of guarded; the clamp only runs in the final chunk (fc2 tail).
// =========================================================================
__global__ void __launch_bounds__(256)
wmma_gemm_kernel(const unsigned short* __restrict__ A,
                 const float* __restrict__ W,
                 const float* __restrict__ bias,
                 float* __restrict__ out,
                 int M, int N, int K, int Kpad, int ldc) {
  __shared__ __align__(16) unsigned short sha[2][16 * KCHUNK];

  int tid  = threadIdx.x;
  int wave = tid >> 5;          // 0..7 : which N-tile of the strip
  int lane = tid & 31;
  int l    = lane & 15;
  int lh   = lane >> 4;
  int m0   = blockIdx.y << 4;
  int n0   = (blockIdx.x << 7) + (wave << 4);

  int  nr  = n0 + l;
  bool nok = nr < N;
  const float* wrow = W + (size_t)(nok ? nr : (N - 1)) * K;  // clamp: always in-range

  // staging assignment: threads 0..191 (6 full waves) copy 16B each per chunk
  int srow = tid / 12;          // 0..15
  int sseg = tid % 12;          // 0..11 (12 * 8 ushorts = 96)
  const unsigned short* gsrc = A + (size_t)(m0 + srow) * Kpad + sseg * 8;

  v8f acc = (v8f){0.f, 0.f, 0.f, 0.f, 0.f, 0.f, 0.f, 0.f};

  int nchunks = Kpad / KCHUNK;  // 768 -> 8, 3360 -> 35 (exact)

  // prefetch chunk 0 into buffer 0 (async DMA to LDS, tracked by ASYNCcnt)
  if (tid < 192) {
    unsigned ldsb = (unsigned)(size_t)&sha[0][srow * KCHUNK + sseg * 8];
    asm volatile("global_load_async_to_lds_b128 %0, %1, off"
                 :: "v"(ldsb), "v"(gsrc) : "memory");
  }

  for (int c = 0; c < nchunks; ++c) {
    asm volatile("s_wait_asynccnt 0x0" ::: "memory");
    __syncthreads();            // tile c published in sha[c&1]

    if (c + 1 < nchunks && tid < 192) {
      unsigned ldsb = (unsigned)(size_t)&sha[(c + 1) & 1][srow * KCHUNK + sseg * 8];
      asm volatile("global_load_async_to_lds_b128 %0, %1, off"
                   :: "v"(ldsb), "v"(gsrc + (size_t)(c + 1) * KCHUNK) : "memory");
    }

    const unsigned short* abase = &sha[c & 1][l * KCHUNK];
    int  kglob = c * KCHUNK;
    bool tail  = (kglob + KCHUNK > K);   // only final chunk of fc2

#pragma unroll
    for (int s = 0; s < 3; ++s) {
      int kc = s * 32;
      // A k order (contiguous, same as B; not ISA table order): lanes 0-15 halves0..15 = K kc..kc+15
      //                            lanes16-31 halves0..15 = kc+16..kc+31
      v16u au;
#pragma unroll
      for (int t = 0; t < 8; ++t) au[t]     = abase[kc + lh * 16 + t];
#pragma unroll
      for (int t = 0; t < 8; ++t) au[8 + t] = abase[kc + lh * 16 + 8 + t];
      v16bf a = __builtin_bit_cast(v16bf, au);

      // B k order (contiguous, same as A): lanes 0-15 hold K kc..kc+15 of col n, lanes 16-31 kc+16..kc+31
      int kb = kglob + kc + lh * 16;
      v16u bu;
      if (!tail) {
#pragma unroll
        for (int t = 0; t < 16; ++t) bu[t] = f2bf_u(wrow[kb + t]);
      } else {
#pragma unroll
        for (int t = 0; t < 16; ++t) {
          int kk = kb + t;
          if (kk >= K) kk = K - 1;         // in-range read; A is zero there anyway
          bu[t] = f2bf_u(wrow[kk]);
        }
      }
      v16bf bm = __builtin_bit_cast(v16bf, bu);
      acc = __builtin_amdgcn_wmma_f32_16x16x32_bf16(false, a, false, bm,
                                                    (short)0, acc, false, false);
    }
    __syncthreads();            // all waves done reading sha[c&1] before re-fill
  }

  // store: VGPR r -> lanes 0-15: (m0+r, n), lanes 16-31: (m0+8+r, n)
  if (nok) {
    float bv = bias ? bias[nr] : 0.0f;
    int   mb = m0 + lh * 8;
#pragma unroll
    for (int r = 0; r < 8; ++r) {
      int m = mb + r;
      if (m < M) out[(size_t)m * ldc + nr] = acc[r] + bv;
    }
  }
}

// =========================================================================
// Patch-embed conv: pe[b, n, c] = sum_{i,d,h,w} x[b,d,i, ...] * w[c,i,d,h,w] + b[c]
// x: (4, 2, 3, 256, 256), w: (768, 3, 2, 16, 16), stride (2,16,16)
// =========================================================================
__global__ void patch_conv_kernel(const float* __restrict__ x,
                                  const float* __restrict__ w,
                                  const float* __restrict__ bias,
                                  float* __restrict__ pe) {
  int idx = blockIdx.x * blockDim.x + threadIdx.x;
  if (idx >= BATCH * 256 * CDIM) return;
  int c = idx % CDIM;
  int n = (idx / CDIM) & 255;
  int b = idx / (CDIM * 256);
  int py = n >> 4, px = n & 15;

  float s = bias[c];
  for (int i = 0; i < 3; ++i) {
    for (int d = 0; d < 2; ++d) {
      const float* xr = x + ((((size_t)b * 2 + d) * 3 + i) * 256 + py * 16) * 256 + px * 16;
      const float* wr = w + (((size_t)c * 3 + i) * 2 + d) * 256;
      for (int hh = 0; hh < 16; ++hh) {
#pragma unroll
        for (int ww = 0; ww < 16; ++ww)
          s += xr[hh * 256 + ww] * wr[hh * 16 + ww];
      }
    }
  }
  pe[((size_t)b * 256 + n) * CDIM + c] = s;
}

// =========================================================================
// Assemble h: token rows = LN(pe) + t_pos[0] + s_pos[n]; row 256 = ar token
// =========================================================================
__global__ void assemble_kernel(const float* __restrict__ pe,
                                const float* __restrict__ lnw,
                                const float* __restrict__ lnb,
                                const float* __restrict__ ar_tok,
                                const float* __restrict__ ar,
                                const float* __restrict__ t_pos,
                                const float* __restrict__ s_pos,
                                float* __restrict__ h) {
  __shared__ float red[256];
  int row = blockIdx.x;          // 0..MROWS-1
  int t   = threadIdx.x;
  int b   = row / SEQ, tok = row % SEQ;
  float* hr = h + (size_t)row * CDIM;

  if (tok == SEQ - 1) {
    float sc = 1.0f + ar[b] * 0.1f;
    hr[t]       = ar_tok[t] * sc;
    hr[t + 256] = ar_tok[t + 256] * sc;
    hr[t + 512] = ar_tok[t + 512] * sc;
    return;
  }
  const float* r  = pe + ((size_t)b * 256 + tok) * CDIM;
  float v0 = r[t], v1 = r[t + 256], v2 = r[t + 512];
  float mean = block_sum256(v0 + v1 + v2, red) * (1.0f / CDIM);
  float d0 = v0 - mean, d1 = v1 - mean, d2 = v2 - mean;
  float var  = block_sum256(d0 * d0 + d1 * d1 + d2 * d2, red) * (1.0f / CDIM);
  float rstd = rsqrtf(var + 1e-5f);
  const float* sp = s_pos + (size_t)tok * CDIM;
  hr[t]       = d0 * rstd * lnw[t]       + lnb[t]       + t_pos[t]       + sp[t];
  hr[t + 256] = d1 * rstd * lnw[t + 256] + lnb[t + 256] + t_pos[t + 256] + sp[t + 256];
  hr[t + 512] = d2 * rstd * lnw[t + 512] + lnb[t + 512] + t_pos[t + 512] + sp[t + 512];
}

// =========================================================================
// LayerNorm -> bf16 activation buffer (Mpad rows, pad rows zeroed)
// =========================================================================
__global__ void ln_bf16_kernel(const float* __restrict__ in,
                               const float* __restrict__ w,
                               const float* __restrict__ b,
                               unsigned short* __restrict__ out,
                               int M, int valid_tok) {
  __shared__ float red[256];
  int row = blockIdx.x;
  int t   = threadIdx.x;
  bool valid = (row < M) && ((row % SEQ) < valid_tok);
  float v0 = 0.f, v1 = 0.f, v2 = 0.f;
  const float* r = in + (size_t)row * CDIM;
  if (valid) { v0 = r[t]; v1 = r[t + 256]; v2 = r[t + 512]; }
  float mean = block_sum256(v0 + v1 + v2, red) * (1.0f / CDIM);
  float d0 = v0 - mean, d1 = v1 - mean, d2 = v2 - mean;
  float var  = block_sum256(d0 * d0 + d1 * d1 + d2 * d2, red) * (1.0f / CDIM);
  float rstd = rsqrtf(var + 1e-5f);
  unsigned short* o = out + (size_t)row * CDIM;
  if (valid) {
    o[t]       = f2bf_u(d0 * rstd * w[t]       + b[t]);
    o[t + 256] = f2bf_u(d1 * rstd * w[t + 256] + b[t + 256]);
    o[t + 512] = f2bf_u(d2 * rstd * w[t + 512] + b[t + 512]);
  } else {
    o[t] = 0; o[t + 256] = 0; o[t + 512] = 0;
  }
}

// =========================================================================
// Final LayerNorm -> fp32 d_out
// =========================================================================
__global__ void final_ln_kernel(const float* __restrict__ in,
                                const float* __restrict__ w,
                                const float* __restrict__ b,
                                float* __restrict__ out) {
  __shared__ float red[256];
  int row = blockIdx.x;
  int t   = threadIdx.x;
  const float* r = in + (size_t)row * CDIM;
  float v0 = r[t], v1 = r[t + 256], v2 = r[t + 512];
  float mean = block_sum256(v0 + v1 + v2, red) * (1.0f / CDIM);
  float d0 = v0 - mean, d1 = v1 - mean, d2 = v2 - mean;
  float var  = block_sum256(d0 * d0 + d1 * d1 + d2 * d2, red) * (1.0f / CDIM);
  float rstd = rsqrtf(var + 1e-5f);
  float* o = out + (size_t)row * CDIM;
  o[t]       = d0 * rstd * w[t]       + b[t];
  o[t + 256] = d1 * rstd * w[t + 256] + b[t + 256];
  o[t + 512] = d2 * rstd * w[t + 512] + b[t + 512];
}

// =========================================================================
// Local 7x7 window attention (zero-padded K/V, matching jnp.pad semantics).
// One wave per (b, head, pixel). hd=64 -> 2 dims per lane.
// =========================================================================
__global__ void local_attn_kernel(const float* __restrict__ qkv,
                                  float* __restrict__ o) {
  int gid  = blockIdx.x;               // b*12*256 + head*256 + pix
  int pix  = gid & 255;
  int head = (gid >> 8) % NHEADS;
  int b    = gid / (256 * NHEADS);
  int y = pix >> 4, x = pix & 15;
  int lane = threadIdx.x;
  int d0 = lane * 2, d1 = d0 + 1;

  const float* qr = qkv + (size_t)(b * SEQ + pix) * C3 + head * HD;
  float q0 = qr[d0], q1 = qr[d1];

  float sc[49];
  float smax = -1.0e30f;
  int j = 0;
  for (int dy = -3; dy <= 3; ++dy) {
    for (int dx = -3; dx <= 3; ++dx, ++j) {
      int yy = y + dy, xx = x + dx;
      float s = 0.0f;
      if (yy >= 0 && yy < 16 && xx >= 0 && xx < 16) {
        const float* kr = qkv + (size_t)(b * SEQ + (yy * 16 + xx)) * C3 + CDIM + head * HD;
        s = wave_sum32(q0 * kr[d0] + q1 * kr[d1]) * 0.125f;
      }
      sc[j] = s;
      smax  = fmaxf(smax, s);
    }
  }

  float denom = 0.f, a0 = 0.f, a1 = 0.f;
  j = 0;
  for (int dy = -3; dy <= 3; ++dy) {
    for (int dx = -3; dx <= 3; ++dx, ++j) {
      float e = __expf(sc[j] - smax);
      denom += e;
      int yy = y + dy, xx = x + dx;
      if (yy >= 0 && yy < 16 && xx >= 0 && xx < 16) {
        const float* vr = qkv + (size_t)(b * SEQ + (yy * 16 + xx)) * C3 + 2 * CDIM + head * HD;
        a0 += e * vr[d0];
        a1 += e * vr[d1];
      }
    }
  }
  float inv = 1.0f / denom;
  float* orow = o + (size_t)(b * SEQ + pix) * CDIM + head * HD;
  orow[d0] = a0 * inv;
  orow[d1] = a1 * inv;
}

// =========================================================================
// Global MHA, flash-style online softmax. One wave per (b, head, query).
// =========================================================================
__global__ void mha_kernel(const float* __restrict__ qkv,
                           float* __restrict__ o) {
  int gid  = blockIdx.x;               // b*12*257 + head*257 + qt
  int qt   = gid % SEQ;
  int head = (gid / SEQ) % NHEADS;
  int b    = gid / (SEQ * NHEADS);
  int lane = threadIdx.x;
  int d0 = lane * 2, d1 = d0 + 1;

  const float* qr = qkv + (size_t)(b * SEQ + qt) * C3 + head * HD;
  float q0 = qr[d0], q1 = qr[d1];

  float m = -3.0e38f, l = 0.f, a0 = 0.f, a1 = 0.f;
  for (int kt = 0; kt < SEQ; ++kt) {
    const float* kr = qkv + (size_t)(b * SEQ + kt) * C3 + CDIM + head * HD;
    float s    = wave_sum32(q0 * kr[d0] + q1 * kr[d1]) * 0.125f;
    float mn   = fmaxf(m, s);
    float corr = __expf(m - mn);
    float e    = __expf(s - mn);
    const float* vr = kr + CDIM;
    l  = l * corr + e;
    a0 = a0 * corr + e * vr[d0];
    a1 = a1 * corr + e * vr[d1];
    m  = mn;
  }
  float inv = 1.0f / l;
  float* orow = o + (size_t)(b * SEQ + qt) * CDIM + head * HD;
  orow[d0] = a0 * inv;
  orow[d1] = a1 * inv;
}

// =========================================================================
// fp32 -> bf16 convert with padding/validity masking
// =========================================================================
__global__ void cvt_bf16_kernel(const float* __restrict__ in,
                                unsigned short* __restrict__ out,
                                int M, int ldin, int Kvalid, int Kpad,
                                int valid_tok, long total) {
  long idx = (long)blockIdx.x * 256 + threadIdx.x;
  if (idx >= total) return;
  int col = (int)(idx % Kpad);
  int row = (int)(idx / Kpad);
  bool valid = (row < M) && (col < Kvalid) && ((row % SEQ) < valid_tok);
  out[idx] = valid ? f2bf_u(in[(size_t)row * ldin + col]) : (unsigned short)0;
}

// =========================================================================
// GELU(exact erf) + fp32 -> bf16 convert with padding
// =========================================================================
__global__ void gelu_bf16_kernel(const float* __restrict__ in,
                                 unsigned short* __restrict__ out,
                                 int M, int ldin, int Kvalid, int Kpad,
                                 int valid_tok, long total) {
  long idx = (long)blockIdx.x * 256 + threadIdx.x;
  if (idx >= total) return;
  int col = (int)(idx % Kpad);
  int row = (int)(idx / Kpad);
  bool valid = (row < M) && (col < Kvalid) && ((row % SEQ) < valid_tok);
  out[idx] = valid ? f2bf_u(gelu_exact(in[(size_t)row * ldin + col])) : (unsigned short)0;
}

// =========================================================================
// Residual add (optionally skipping the aspect-ratio token row)
// =========================================================================
__global__ void residual_add_kernel(float* __restrict__ h,
                                    const float* __restrict__ src,
                                    int valid_tok, int total) {
  int idx = blockIdx.x * blockDim.x + threadIdx.x;
  if (idx >= total) return;
  int row = idx / CDIM;
  if ((row % SEQ) < valid_tok) h[idx] += src[idx];
}

// =========================================================================
// Host orchestration
// =========================================================================
extern "C" void kernel_launch(void* const* d_in, const int* in_sizes, int n_in,
                              void* d_out, int out_size, void* d_ws, size_t ws_size,
                              hipStream_t stream) {
  (void)in_sizes; (void)n_in; (void)out_size; (void)ws_size;

  const float* x         = (const float*)d_in[0];
  const float* aspect    = (const float*)d_in[1];
  const float* conv_w    = (const float*)d_in[2];
  const float* conv_b    = (const float*)d_in[3];
  const float* pe_ln_w   = (const float*)d_in[4];
  const float* pe_ln_b   = (const float*)d_in[5];
  const float* ar_tok    = (const float*)d_in[6];
  const float* t_pos     = (const float*)d_in[7];
  const float* s_pos     = (const float*)d_in[8];
  const float* n1_w      = (const float*)d_in[9];
  const float* n1_b      = (const float*)d_in[10];
  const float* n2_w      = (const float*)d_in[11];
  const float* n2_b      = (const float*)d_in[12];
  const float* loc_qkv_w = (const float*)d_in[13];
  const float* loc_proj_w= (const float*)d_in[14];
  const float* loc_proj_b= (const float*)d_in[15];
  const float* glb_in_w  = (const float*)d_in[16];
  const float* glb_in_b  = (const float*)d_in[17];
  const float* glb_out_w = (const float*)d_in[18];
  const float* glb_out_b = (const float*)d_in[19];
  const float* fc1_w     = (const float*)d_in[20];
  const float* fc1_b     = (const float*)d_in[21];
  const float* fc2_w     = (const float*)d_in[22];
  const float* fc2_b     = (const float*)d_in[23];
  const float* fn_w      = (const float*)d_in[24];
  const float* fn_b      = (const float*)d_in[25];

  char*  wsp = (char*)d_ws;
  size_t off = 0;
  auto alloc = [&](size_t bytes) -> char* {
    char* p = wsp + off;
    off = (off + bytes + 255) & ~(size_t)255;
    return p;
  };

  float*          pe     = (float*)alloc((size_t)BATCH * 256 * CDIM * 4);
  float*          h      = (float*)alloc((size_t)MROWS * CDIM * 4);
  unsigned short* actbf  = (unsigned short*)alloc((size_t)MPAD * CDIM * 2);
  float*          qkv    = (float*)alloc((size_t)MROWS * C3 * 4);
  float*          attno  = (float*)alloc((size_t)MROWS * CDIM * 4);
  unsigned short* attnbf = (unsigned short*)alloc((size_t)MPAD * CDIM * 2);
  float*          proj   = (float*)alloc((size_t)MROWS * CDIM * 4);
  float*          fc1o   = (float*)alloc((size_t)MROWS * CHP * 4);
  unsigned short* gelubf = (unsigned short*)alloc((size_t)MPAD * CHP * 2);

  // 1. patch embed + token assembly
  int nconv = BATCH * 256 * CDIM;
  patch_conv_kernel<<<(nconv + 255) / 256, 256, 0, stream>>>(x, conv_w, conv_b, pe);
  assemble_kernel<<<MROWS, 256, 0, stream>>>(pe, pe_ln_w, pe_ln_b, ar_tok, aspect,
                                             t_pos, s_pos, h);

  const int  addTotal = MROWS * CDIM;
  const long cvtTot   = (long)MPAD * CDIM;
  const long geluTot  = (long)MPAD * CHP;
  const int  mtiles   = MPAD / 16;   // 65

  for (int i = 0; i < 12; ++i) {
    if ((i & 1) == 0) {
      // ---- local window attention ----
      int e = i >> 1;
      ln_bf16_kernel<<<MPAD, 256, 0, stream>>>(h, n1_w + i * CDIM, n1_b + i * CDIM,
                                               actbf, MROWS, SEQ);
      dim3 g1((C3 + 127) / 128, mtiles);
      wmma_gemm_kernel<<<g1, 256, 0, stream>>>(actbf, loc_qkv_w + (size_t)e * C3 * CDIM,
                                               nullptr, qkv, MROWS, C3, CDIM, CDIM, C3);
      local_attn_kernel<<<BATCH * NHEADS * 256, 32, 0, stream>>>(qkv, attno);
      cvt_bf16_kernel<<<(unsigned)((cvtTot + 255) / 256), 256, 0, stream>>>(
          attno, attnbf, MROWS, CDIM, CDIM, CDIM, SEQ - 1, cvtTot);
      dim3 g2((CDIM + 127) / 128, mtiles);
      wmma_gemm_kernel<<<g2, 256, 0, stream>>>(attnbf, loc_proj_w + (size_t)e * CDIM * CDIM,
                                               loc_proj_b + e * CDIM, proj,
                                               MROWS, CDIM, CDIM, CDIM, CDIM);
      residual_add_kernel<<<(addTotal + 255) / 256, 256, 0, stream>>>(h, proj, SEQ - 1, addTotal);
    } else {
      // ---- global MHA ----
      int g = i >> 1;
      ln_bf16_kernel<<<MPAD, 256, 0, stream>>>(h, n1_w + i * CDIM, n1_b + i * CDIM,
                                               actbf, MROWS, SEQ);
      dim3 g1((C3 + 127) / 128, mtiles);
      wmma_gemm_kernel<<<g1, 256, 0, stream>>>(actbf, glb_in_w + (size_t)g * C3 * CDIM,
                                               glb_in_b + g * C3, qkv,
                                               MROWS, C3, CDIM, CDIM, C3);
      mha_kernel<<<BATCH * NHEADS * SEQ, 32, 0, stream>>>(qkv, attno);
      cvt_bf16_kernel<<<(unsigned)((cvtTot + 255) / 256), 256, 0, stream>>>(
          attno, attnbf, MROWS, CDIM, CDIM, CDIM, SEQ, cvtTot);
      dim3 g2((CDIM + 127) / 128, mtiles);
      wmma_gemm_kernel<<<g2, 256, 0, stream>>>(attnbf, glb_out_w + (size_t)g * CDIM * CDIM,
                                               glb_out_b + g * CDIM, proj,
                                               MROWS, CDIM, CDIM, CDIM, CDIM);
      residual_add_kernel<<<(addTotal + 255) / 256, 256, 0, stream>>>(h, proj, SEQ, addTotal);
    }

    // ---- MLP ----
    ln_bf16_kernel<<<MPAD, 256, 0, stream>>>(h, n2_w + i * CDIM, n2_b + i * CDIM,
                                             actbf, MROWS, SEQ);
    dim3 g3((CHP + 127) / 128, mtiles);
    wmma_gemm_kernel<<<g3, 256, 0, stream>>>(actbf, fc1_w + (size_t)i * CH * CDIM,
                                             fc1_b + i * CH, fc1o,
                                             MROWS, CH, CDIM, CDIM, CHP);
    gelu_bf16_kernel<<<(unsigned)((geluTot + 255) / 256), 256, 0, stream>>>(
        fc1o, gelubf, MROWS, CHP, CH, CHP, SEQ, geluTot);
    dim3 g4((CDIM + 127) / 128, mtiles);
    wmma_gemm_kernel<<<g4, 256, 0, stream>>>(gelubf, fc2_w + (size_t)i * CDIM * CH,
                                             fc2_b + i * CDIM, proj,
                                             MROWS, CDIM, CH, CHP, CDIM);
    residual_add_kernel<<<(addTotal + 255) / 256, 256, 0, stream>>>(h, proj, SEQ, addTotal);
  }

  // 3. final LayerNorm -> d_out
  final_ln_kernel<<<MROWS, 256, 0, stream>>>(h, fn_w, fn_b, (float*)d_out);
}
